// NodeBlock_11527692222556
// MI455X (gfx1250) — matrix operand order repaired and checked by reading, not compiled, not run on hardware
//
#include <hip/hip_runtime.h>
#include <hip/hip_bf16.h>

// ---------- types ----------
typedef __attribute__((ext_vector_type(16))) __bf16    v16bf;
typedef __attribute__((ext_vector_type(8)))  float     v8f;
typedef __attribute__((ext_vector_type(4)))  unsigned  v4u;

union Frag { v4u u[2]; v16bf v; };

// ---------- small utility kernels ----------
__global__ __launch_bounds__(256) void zero_f32(float* __restrict__ p, int n) {
    int i = blockIdx.x * 256 + threadIdx.x;
    if (i < n) p[i] = 0.0f;
}

// one thread per (edge, feature): coalesced edge_feat reads, atomic scatter into agg
__global__ __launch_bounds__(256) void scatter_add_edges(
    const float* __restrict__ edge_feat, const int* __restrict__ dst,
    float* __restrict__ agg, int n_edges)
{
    long long i = (long long)blockIdx.x * 256 + threadIdx.x;
    long long total = (long long)n_edges * 96;
    if (i >= total) return;
    int e = (int)(i / 96);
    int f = (int)(i % 96);
    atomicAdd(&agg[(size_t)dst[e] * 96 + f], edge_feat[i]);
}

// build bf16 input matrix [M x 608] = concat(node_feat[Mx512], agg[Mx96])
__global__ __launch_bounds__(256) void build_input_bf16(
    const float* __restrict__ node, const float* __restrict__ agg,
    __bf16* __restrict__ A, int M)
{
    long long i = (long long)blockIdx.x * 256 + threadIdx.x;
    long long total = (long long)M * 608;
    if (i >= total) return;
    int m = (int)(i / 608);
    int c = (int)(i % 608);
    float v = (c < 512) ? node[(size_t)m * 512 + c] : agg[(size_t)m * 96 + (c - 512)];
    A[i] = (__bf16)v;
}

// W [K x N] fp32 row-major  ->  Wt [N x K] bf16 row-major (B transposed for WMMA B-frag loads)
__global__ __launch_bounds__(256) void transpose_convert_bf16(
    const float* __restrict__ W, __bf16* __restrict__ Wt, int K, int N)
{
    long long i = (long long)blockIdx.x * 256 + threadIdx.x;
    long long total = (long long)K * N;
    if (i >= total) return;
    int k = (int)(i / N);
    int n = (int)(i % N);
    Wt[(size_t)n * K + k] = (__bf16)W[i];
}

// ---------- WMMA GEMM: C[MxN] = act(A[MxK] @ B[KxN] + bias), B given as Bt[NxK] ----------
// block tile 128x128, 8 waves (4 in M x 2 in N), wave tile 32x64 -> 8 v_wmma per K-chunk(32).
// Double-buffered LDS filled with GLOBAL_LOAD_ASYNC_TO_LDS_B128 (ASYNCcnt), one barrier/iter.
#define LDS_STRIDE 40   // 32 bf16 + 8 pad (80B rows, 16B aligned)

// per-lane async copy of 16B global -> LDS (EXEC must cover the lane)
__device__ __forceinline__ void async_b128_to_lds(unsigned lds_byte_addr, const void* gaddr) {
    asm volatile("global_load_async_to_lds_b128 %0, %1, off"
                 :: "v"(lds_byte_addr), "v"(gaddr) : "memory");
}
__device__ __forceinline__ void wait_async0() {
    asm volatile("s_wait_asynccnt 0x0" ::: "memory");
}
__device__ __forceinline__ unsigned lds_addr_of(const void* p) {
    // LDS aperture: flat address low 32 bits are the LDS byte offset
    return (unsigned)(unsigned long long)p;
}

template <bool RELU, typename OutT>
__global__ __launch_bounds__(256) void gemm_bias_act(
    const __bf16* __restrict__ A,   // [M x K]
    const __bf16* __restrict__ Bt,  // [N x K]
    const float*  __restrict__ bias,// [N]
    OutT*         __restrict__ C,   // [M x N]
    int M, int N, int K)
{
    __shared__ __bf16 As[2][128 * LDS_STRIDE];
    __shared__ __bf16 Bs[2][128 * LDS_STRIDE];

    const int tid   = threadIdx.x;
    const int lane  = tid & 31;
    const int wave  = tid >> 5;         // 0..7
    const int waveM = wave & 3;         // 0..3  -> 32 rows each
    const int waveN = wave >> 2;        // 0..1  -> 64 cols each
    const int l16   = lane & 15;
    const int hi    = (lane >> 4) & 1;  // half-wave select

    const int mBlock = blockIdx.x * 128;
    const int nBlock = blockIdx.y * 128;

    // per-thread tile-load coordinates: 512 x 16B segments per tile, 2 per thread
    // linear = tid + i*256; row = linear>>2 (0..127); seg = linear&3
    int rowA0 = tid >> 2,        segA0 = tid & 3;
    int rowA1 = (tid + 256) >> 2, segA1 = tid & 3;      // (tid+256)&3 == tid&3
    int mA0 = mBlock + rowA0; if (mA0 >= M) mA0 = M - 1;   // clamp: keep EXEC full
    int mA1 = mBlock + rowA1; if (mA1 >= M) mA1 = M - 1;
    const __bf16* gA0 = A + (size_t)mA0 * K + segA0 * 8;
    const __bf16* gA1 = A + (size_t)mA1 * K + segA1 * 8;
    const __bf16* gB0 = Bt + (size_t)(nBlock + rowA0) * K + segA0 * 8;
    const __bf16* gB1 = Bt + (size_t)(nBlock + rowA1) * K + segA1 * 8;

    v8f acc[2][4];
#pragma unroll
    for (int a = 0; a < 2; ++a)
#pragma unroll
        for (int b = 0; b < 4; ++b) acc[a][b] = (v8f)0.0f;

    const int nk = K >> 5;              // K is a multiple of 32 (608, 1024)

    auto issue_tile = [&](int buf, int kc) {
        int ko = kc * 32;
        async_b128_to_lds(lds_addr_of(&As[buf][rowA0 * LDS_STRIDE + segA0 * 8]), gA0 + ko);
        async_b128_to_lds(lds_addr_of(&As[buf][rowA1 * LDS_STRIDE + segA1 * 8]), gA1 + ko);
        async_b128_to_lds(lds_addr_of(&Bs[buf][rowA0 * LDS_STRIDE + segA0 * 8]), gB0 + ko);
        async_b128_to_lds(lds_addr_of(&Bs[buf][rowA1 * LDS_STRIDE + segA1 * 8]), gB1 + ko);
    };

    issue_tile(0, 0);

    for (int kc = 0; kc < nk; ++kc) {
        const int cur = kc & 1;
        wait_async0();          // this buffer's async copies (issued last iter) done
        __syncthreads();        // all waves' copies done -> LDS tile coherent;
                                // also: everyone finished reading buffer 'cur^1'
        if (kc + 1 < nk) issue_tile(cur ^ 1, kc + 1);

        // -------- fragments per CDNA5 16-bit WMMA VGPR layouts --------
        Frag af[2], bfr[4];
#pragma unroll
        for (int m16 = 0; m16 < 2; ++m16) {
            int r    = waveM * 32 + m16 * 16 + l16;      // A row
            int kOff = hi ? 8 : 0;                        // lanes16-31: K 8..15 / 24..31
            af[m16].u[0] = *reinterpret_cast<const v4u*>(&As[cur][r * LDS_STRIDE + kOff]);
            af[m16].u[1] = *reinterpret_cast<const v4u*>(&As[cur][r * LDS_STRIDE + kOff + 16]);
        }
#pragma unroll
        for (int n16 = 0; n16 < 4; ++n16) {
            int c  = waveN * 64 + n16 * 16 + l16;        // B column
            int kB = hi ? 8 : 0;                          // lanes16-31: K 8..15 / 24..31
            bfr[n16].u[0] = *reinterpret_cast<const v4u*>(&Bs[cur][c * LDS_STRIDE + kB]);
            bfr[n16].u[1] = *reinterpret_cast<const v4u*>(&Bs[cur][c * LDS_STRIDE + kB + 16]);
        }

#pragma unroll
        for (int m16 = 0; m16 < 2; ++m16)
#pragma unroll
            for (int n16 = 0; n16 < 4; ++n16)
                acc[m16][n16] = __builtin_amdgcn_wmma_f32_16x16x32_bf16(
                    false, af[m16].v, false, bfr[n16].v,
                    (short)0, acc[m16][n16], false, false);
    }

    // -------- epilogue: bias (+ReLU), convert, store --------
#pragma unroll
    for (int m16 = 0; m16 < 2; ++m16) {
#pragma unroll
        for (int n16 = 0; n16 < 4; ++n16) {
            int col = nBlock + waveN * 64 + n16 * 16 + l16;
            float bb = bias[col];
#pragma unroll
            for (int r = 0; r < 8; ++r) {
                int row = mBlock + waveM * 32 + m16 * 16 + hi * 8 + r;
                if (row < M) {
                    float v = acc[m16][n16][r] + bb;
                    if (RELU) v = fmaxf(v, 0.0f);
                    C[(size_t)row * N + col] = (OutT)v;
                }
            }
        }
    }
}

// ---------- per-row LayerNorm, in place on fp32 [M x 512] ----------
__global__ __launch_bounds__(256) void layernorm_inplace(
    float* __restrict__ y, const float* __restrict__ gamma,
    const float* __restrict__ beta)
{
    __shared__ float s1[256];
    __shared__ float s2[256];
    const int t = threadIdx.x;
    float* p = y + (size_t)blockIdx.x * 512;
    float a = p[t];
    float b = p[t + 256];
    s1[t] = a + b;
    s2[t] = a * a + b * b;
    __syncthreads();
#pragma unroll
    for (int off = 128; off > 0; off >>= 1) {
        if (t < off) { s1[t] += s1[t + off]; s2[t] += s2[t + off]; }
        __syncthreads();
    }
    float mean = s1[0] * (1.0f / 512.0f);
    float var  = s2[0] * (1.0f / 512.0f) - mean * mean;
    float inv  = rsqrtf(var + 1e-5f);
    p[t]       = (a - mean) * inv * gamma[t]       + beta[t];
    p[t + 256] = (b - mean) * inv * gamma[t + 256] + beta[t + 256];
}

// ---------- host launch ----------
extern "C" void kernel_launch(void* const* d_in, const int* in_sizes, int n_in,
                              void* d_out, int out_size, void* d_ws, size_t ws_size,
                              hipStream_t stream)
{
    const float* node_feat = (const float*)d_in[0];
    const float* edge_feat = (const float*)d_in[1];
    const int*   edge_idx  = (const int*)d_in[2];
    const float* W1 = (const float*)d_in[4];
    const float* b1 = (const float*)d_in[5];
    const float* W2 = (const float*)d_in[6];
    const float* b2 = (const float*)d_in[7];
    const float* W3 = (const float*)d_in[8];
    const float* b3 = (const float*)d_in[9];
    const float* gamma = (const float*)d_in[10];
    const float* beta  = (const float*)d_in[11];

    const int M  = in_sizes[0] / 512;   // nodes
    const int E  = in_sizes[1] / 96;    // edges
    const int IN = 608, H = 1024, OUT = 512;

    // workspace carve-out (256B aligned)
    size_t off = 0;
    auto carve = [&](size_t bytes) {
        size_t o = off;
        off = (off + bytes + 255) & ~(size_t)255;
        return o;
    };
    char* ws = (char*)d_ws;
    float*  agg  = (float*)(ws + carve((size_t)M * 96 * sizeof(float)));
    __bf16* Abf  = (__bf16*)(ws + carve((size_t)M * IN * sizeof(__bf16)));
    __bf16* W1t  = (__bf16*)(ws + carve((size_t)H * IN * sizeof(__bf16)));
    __bf16* W2t  = (__bf16*)(ws + carve((size_t)H * H * sizeof(__bf16)));
    __bf16* W3t  = (__bf16*)(ws + carve((size_t)OUT * H * sizeof(__bf16)));
    __bf16* h1   = (__bf16*)(ws + carve((size_t)M * H * sizeof(__bf16)));
    __bf16* h2   = (__bf16*)(ws + carve((size_t)M * H * sizeof(__bf16)));
    float* y = (float*)d_out;

    // 1) zero aggregation buffer
    {
        int n = M * 96;
        zero_f32<<<(n + 255) / 256, 256, 0, stream>>>(agg, n);
    }
    // 2) scatter-add incoming edge features (dst = edge_index row 1)
    {
        long long n = (long long)E * 96;
        int blocks = (int)((n + 255) / 256);
        scatter_add_edges<<<blocks, 256, 0, stream>>>(edge_feat, edge_idx + E, agg, E);
    }
    // 3) build bf16 concatenated input [M x 608]
    {
        long long n = (long long)M * IN;
        build_input_bf16<<<(int)((n + 255) / 256), 256, 0, stream>>>(node_feat, agg, Abf, M);
    }
    // 4) transpose + convert weights to bf16 [N x K]
    {
        long long n;
        n = (long long)IN * H;
        transpose_convert_bf16<<<(int)((n + 255) / 256), 256, 0, stream>>>(W1, W1t, IN, H);
        n = (long long)H * H;
        transpose_convert_bf16<<<(int)((n + 255) / 256), 256, 0, stream>>>(W2, W2t, H, H);
        n = (long long)H * OUT;
        transpose_convert_bf16<<<(int)((n + 255) / 256), 256, 0, stream>>>(W3, W3t, H, OUT);
    }
    // 5) three WMMA GEMMs
    const int mBlocks = (M + 127) / 128;
    gemm_bias_act<true,  __bf16><<<dim3(mBlocks, H / 128), 256, 0, stream>>>(
        Abf, W1t, b1, h1, M, H, IN);
    gemm_bias_act<true,  __bf16><<<dim3(mBlocks, H / 128), 256, 0, stream>>>(
        h1, W2t, b2, h2, M, H, H);
    gemm_bias_act<false, float ><<<dim3(mBlocks, OUT / 128), 256, 0, stream>>>(
        h2, W3t, b3, y, M, OUT, H);
    // 6) LayerNorm in place on d_out
    layernorm_inplace<<<M, 256, 0, stream>>>(y, gamma, beta);
}
